// JNetwork_32976758898961
// MI455X (gfx1250) — matrix operand order repaired and checked by reading, not compiled, not run on hardware
//
#include <hip/hip_runtime.h>
#include <hip/hip_bf16.h>

#define S_DIM   512
#define R_DIM   32768
#define B_DIM   1024
#define M_TILE  32
#define KSPLIT  16
#define K_CHUNK (R_DIM / KSPLIT)   // 2048
#define RSTRIDE 136                // padded bf16 row stride (2-way-min LDS banking)

typedef __attribute__((ext_vector_type(16))) __bf16 v16bf;
typedef __attribute__((ext_vector_type(8)))  __bf16 v8bf;
typedef __attribute__((ext_vector_type(8)))  float  v8f;

__global__ void __launch_bounds__(256)
zero_f32(float* __restrict__ p, int n) {
    int i = blockIdx.x * blockDim.x + threadIdx.x;
    if (i < n) p[i] = 0.0f;
}

// One-time (per call) f32 -> bf16 conversion of incidence into workspace.
// Values are small integers, so this is exact.
__global__ void __launch_bounds__(256)
cvt_inc_bf16(const float* __restrict__ src, __bf16* __restrict__ dst, int n8) {
    int i = blockIdx.x * blockDim.x + threadIdx.x;
    if (i < n8) {
        size_t base = (size_t)i * 8;
        float4 a = *reinterpret_cast<const float4*>(src + base);
        float4 b = *reinterpret_cast<const float4*>(src + base + 4);
        v8bf o;
        o[0] = (__bf16)a.x; o[1] = (__bf16)a.y; o[2] = (__bf16)a.z; o[3] = (__bf16)a.w;
        o[4] = (__bf16)b.x; o[5] = (__bf16)b.y; o[6] = (__bf16)b.z; o[7] = (__bf16)b.w;
        *reinterpret_cast<v8bf*>(dst + base) = o;
    }
}

template <bool USE_WS>
__global__ void __launch_bounds__(256)
jnet_fused(const float* __restrict__ abund,
           const float* __restrict__ temperature,
           const float* __restrict__ cr_rate,
           const float* __restrict__ fuv_rate,
           const float* __restrict__ incidence,   // f32 path
           const __bf16* __restrict__ incw,       // pre-converted bf16 path
           const float* __restrict__ alpha,
           const float* __restrict__ beta,
           const float* __restrict__ gamma_,
           const float* __restrict__ cr_coef,
           const float* __restrict__ fuv_coef,
           const int*   __restrict__ msi,
           float* __restrict__ out)
{
    __shared__ float  s_ab[M_TILE][S_DIM];       // 64 KB abundances tile
    __shared__ __bf16 s_hi[M_TILE][RSTRIDE];     // rates, high bf16 plane
    __shared__ __bf16 s_lo[M_TILE][RSTRIDE];     // rates, low  bf16 plane
    __shared__ float  s_lt[M_TILE], s_iT[M_TILE], s_cr[M_TILE], s_fuv[M_TILE];

    const int tid = threadIdx.x;
    const int b0  = blockIdx.x * M_TILE;
    const int r0  = blockIdx.y * K_CHUNK;

    // stage abundances tile (rows b0..b0+31 are contiguous in global)
    {
        const float4* g = reinterpret_cast<const float4*>(abund + (size_t)b0 * S_DIM);
        float4* l = reinterpret_cast<float4*>(&s_ab[0][0]);
        for (int i = tid; i < M_TILE * S_DIM / 4; i += 256) l[i] = g[i];
    }
    if (tid < M_TILE) {
        float T = temperature[b0 + tid];
        s_lt[tid]  = __logf(T * (1.0f / 300.0f));
        s_iT[tid]  = 1.0f / T;
        s_cr[tid]  = cr_rate[b0 + tid];
        s_fuv[tid] = fuv_rate[b0 + tid];
    }
    __syncthreads();

    const int wave  = tid >> 5;
    const int lane  = tid & 31;
    const int laneN = lane & 15;
    const int half  = lane >> 4;
    const int n0    = wave * 64;          // each of 8 waves owns 64 species cols

    v8f acc[2][4];
    #pragma unroll
    for (int mf = 0; mf < 2; ++mf)
        #pragma unroll
        for (int nf = 0; nf < 4; ++nf)
            acc[mf][nf] = (v8f){0.f,0.f,0.f,0.f,0.f,0.f,0.f,0.f};

    const int rl = tid & 127;             // reaction column within 128-wide panel
    const int bg = (tid >> 7) * 16;       // batch-row group (0 or 16)

    for (int kb = 0; kb < K_CHUNK; kb += 128) {
        const int rbase = r0 + kb;

        // ---- fused rate computation -> split-precision LDS planes ----
        {
            const int r = rbase + rl;
            const float al = alpha[r];
            const float be = beta[r];
            const float ga = gamma_[r];
            const float cc = cr_coef[r];
            const float fc = fuv_coef[r];
            const int s1 = msi[2 * r];
            const int s2 = msi[2 * r + 1];
            #pragma unroll
            for (int j = 0; j < 16; ++j) {
                const int bl = bg + j;
                float e    = __expf(be * s_lt[bl] - ga * s_iT[bl]);
                float rate = (al * e + cc * s_cr[bl] + fc * s_fuv[bl])
                           * s_ab[bl][s1] * s_ab[bl][s2];
                __bf16 h = (__bf16)rate;
                __bf16 l = (__bf16)(rate - (float)h);
                s_hi[bl][rl] = h;
                s_lo[bl][rl] = l;
            }
        }
        __syncthreads();

        // ---- WMMA over 4 K-steps of 32 ----
        #pragma unroll
        for (int ks = 0; ks < 4; ++ks) {
            const int kcol = ks * 32;

            // A fragments (contiguous K layout matching B: lanes0-15 K0-15, lanes16-31 K16-31)
            v16bf aHi[2], aLo[2];
            #pragma unroll
            for (int mf = 0; mf < 2; ++mf) {
                const int row = mf * 16 + laneN;
                const __bf16* ph = &s_hi[row][kcol + half * 16];
                v8bf h0 = *reinterpret_cast<const v8bf*>(ph);
                v8bf h1 = *reinterpret_cast<const v8bf*>(ph + 8);
                aHi[mf] = __builtin_shufflevector(h0, h1,
                          0,1,2,3,4,5,6,7,8,9,10,11,12,13,14,15);
                const __bf16* pl = &s_lo[row][kcol + half * 16];
                v8bf l0 = *reinterpret_cast<const v8bf*>(pl);
                v8bf l1 = *reinterpret_cast<const v8bf*>(pl + 8);
                aLo[mf] = __builtin_shufflevector(l0, l1,
                          0,1,2,3,4,5,6,7,8,9,10,11,12,13,14,15);
            }

            #pragma unroll
            for (int nf = 0; nf < 4; ++nf) {
                // B fragment: B[k][n] = incidence[s=n][r=k]; contiguous per lane (L2-resident)
                const int s = n0 + nf * 16 + laneN;
                v16bf bfrag;
                if (USE_WS) {
                    const __bf16* gp = incw + (size_t)s * R_DIM + rbase + kcol + half * 16;
                    v8bf c0 = *reinterpret_cast<const v8bf*>(gp);
                    v8bf c1 = *reinterpret_cast<const v8bf*>(gp + 8);
                    bfrag = __builtin_shufflevector(c0, c1,
                            0,1,2,3,4,5,6,7,8,9,10,11,12,13,14,15);
                } else {
                    const float4* gp = reinterpret_cast<const float4*>(
                        incidence + (size_t)s * R_DIM + rbase + kcol + half * 16);
                    #pragma unroll
                    for (int q = 0; q < 4; ++q) {
                        float4 f = gp[q];
                        bfrag[4*q+0] = (__bf16)f.x;
                        bfrag[4*q+1] = (__bf16)f.y;
                        bfrag[4*q+2] = (__bf16)f.z;
                        bfrag[4*q+3] = (__bf16)f.w;
                    }
                }
                #pragma unroll
                for (int mf = 0; mf < 2; ++mf) {
                    acc[mf][nf] = __builtin_amdgcn_wmma_f32_16x16x32_bf16(
                        false, aHi[mf], false, bfrag, (short)0, acc[mf][nf], false, false);
                    acc[mf][nf] = __builtin_amdgcn_wmma_f32_16x16x32_bf16(
                        false, aLo[mf], false, bfrag, (short)0, acc[mf][nf], false, false);
                }
            }
        }
        __syncthreads();
    }

    // ---- K is split over blockIdx.y: accumulate with f32 atomics ----
    #pragma unroll
    for (int mf = 0; mf < 2; ++mf) {
        #pragma unroll
        for (int nf = 0; nf < 4; ++nf) {
            const int s = n0 + nf * 16 + laneN;
            #pragma unroll
            for (int i = 0; i < 8; ++i) {
                const int b = b0 + mf * 16 + half * 8 + i;
                atomicAdd(&out[(size_t)b * S_DIM + s], acc[mf][nf][i]);
            }
        }
    }
}

extern "C" void kernel_launch(void* const* d_in, const int* in_sizes, int n_in,
                              void* d_out, int out_size, void* d_ws, size_t ws_size,
                              hipStream_t stream) {
    // setup_inputs order:
    // 0 time, 1 abundances, 2 temperature, 3 cr_rate, 4 fuv_rate, 5 incidence,
    // 6 alpha, 7 beta, 8 gamma, 9 cr_coef, 10 fuv_coef, 11 mult_reac_idx, 12 mult_species_idx
    const float* abund = (const float*)d_in[1];
    const float* temp  = (const float*)d_in[2];
    const float* crr   = (const float*)d_in[3];
    const float* fuvr  = (const float*)d_in[4];
    const float* inc   = (const float*)d_in[5];
    const float* alpha = (const float*)d_in[6];
    const float* beta  = (const float*)d_in[7];
    const float* gam   = (const float*)d_in[8];
    const float* crc   = (const float*)d_in[9];
    const float* fuvc  = (const float*)d_in[10];
    const int*   msi   = (const int*)d_in[12];
    float* out = (float*)d_out;

    const int n = B_DIM * S_DIM;
    zero_f32<<<(n + 255) / 256, 256, 0, stream>>>(out, n);

    dim3 grid(B_DIM / M_TILE, KSPLIT);   // 32 x 16 = 512 workgroups

    const size_t ws_needed = (size_t)S_DIM * R_DIM * 2;  // bf16 incidence copy
    if (ws_size >= ws_needed) {
        __bf16* incw = (__bf16*)d_ws;
        const int n8 = S_DIM * R_DIM / 8;
        cvt_inc_bf16<<<(n8 + 255) / 256, 256, 0, stream>>>(inc, incw, n8);
        jnet_fused<true><<<grid, 256, 0, stream>>>(abund, temp, crr, fuvr,
                                                   inc, incw,
                                                   alpha, beta, gam, crc, fuvc, msi, out);
    } else {
        jnet_fused<false><<<grid, 256, 0, stream>>>(abund, temp, crr, fuvr,
                                                    inc, (const __bf16*)nullptr,
                                                    alpha, beta, gam, crc, fuvc, msi, out);
    }
}
